// KvvGaussianLayer_51805895524663
// MI455X (gfx1250) — matrix operand order repaired and checked by reading, not compiled, not run on hardware
//
#include <hip/hip_runtime.h>
#include <hip/hip_bf16.h>

typedef _Float16 v16h __attribute__((ext_vector_type(16)));
typedef _Float16 v8h  __attribute__((ext_vector_type(8)));
typedef float    v8f  __attribute__((ext_vector_type(8)));

#define B_   8
#define T_   2048
#define D_   64
#define NF   67
#define JITTER_ 1e-6f
#define TILE 128
#define LDS_STRIDE 72   // halves; 144B row stride -> conflict-free 16B fragment reads

// ---------------------------------------------------------------------------
// Prep: per (b,t) row -> mean (to output), z/8 as f16 (ws), sq/v/softplus (ws)
// ---------------------------------------------------------------------------
__global__ __launch_bounds__(256)
void kvv_prep_kernel(const float* __restrict__ in,
                     float* __restrict__ outMean,
                     _Float16* __restrict__ zh,
                     float* __restrict__ sqv,
                     float* __restrict__ vv,
                     float* __restrict__ nv) {
  int r = blockIdx.x * blockDim.x + threadIdx.x;   // row in [0, B*T)
  if (r >= B_ * T_) return;
  const float* row = in + (size_t)r * NF;
  outMean[r] = row[0];
  float s = 0.f;
  _Float16* zr = zh + (size_t)r * D_;
#pragma unroll 8
  for (int d = 0; d < D_; ++d) {
    float x = row[1 + d] * 0.125f;   // z / sqrt(64)
    s += x * x;
    zr[d] = (_Float16)x;
  }
  sqv[r] = s;
  vv[r] = row[1 + D_];
  float xn = row[2 + D_];            // stable softplus
  nv[r] = (xn > 0.f) ? (xn + log1pf(__expf(-xn))) : log1pf(__expf(xn));
}

// ---------------------------------------------------------------------------
// Cov: one 128x128 output tile per workgroup; 8 waves, each 32x64 via WMMA
// ---------------------------------------------------------------------------
__global__ __launch_bounds__(256)
void kvv_cov_kernel(const _Float16* __restrict__ zh,
                    const float* __restrict__ sqv,
                    const float* __restrict__ vv,
                    const float* __restrict__ nv,
                    float* __restrict__ outF,
                    float* __restrict__ outY) {
  __shared__ _Float16 ldsA[TILE * LDS_STRIDE];
  __shared__ _Float16 ldsB[TILE * LDS_STRIDE];

  const int tj = blockIdx.x, ti = blockIdx.y, b = blockIdx.z;
  const int tid  = threadIdx.x;
  const int lane = tid & 31, wave = tid >> 5;
  const int bT   = b * T_;
  const int rowI = ti * TILE, rowJ = tj * TILE;

  // ---- stage both 128x64 f16 tiles into LDS (16B chunks, coalesced) ----
  const _Float16* zb = zh + (size_t)bT * D_;
#pragma unroll
  for (int p = 0; p < 4; ++p) {
    int idx = tid + p * 256;     // 0..1023
    int r = idx >> 3;            // 0..127
    int c = idx & 7;             // 8-half chunk within row
    uint4 va = *(const uint4*)(zb + (size_t)(rowI + r) * D_ + c * 8);
    *(uint4*)&ldsA[r * LDS_STRIDE + c * 8] = va;
    uint4 vb = *(const uint4*)(zb + (size_t)(rowJ + r) * D_ + c * 8);
    *(uint4*)&ldsB[r * LDS_STRIDE + c * 8] = vb;
  }
  __syncthreads();

  const int wr   = wave >> 1;    // 0..3 : 32-row band
  const int wc   = wave & 1;     // 0..1 : 64-col band
  const int half = lane >> 4;    // lane half (ISA fragment layouts)
  const int l16  = lane & 15;

  v8f acc[2][4] = {};

#pragma unroll
  for (int ks = 0; ks < 2; ++ks) {
    const int k0 = ks * 32;

    // A fragments: lane<16 holds M=l16, K in [k0,k0+8)∪[k0+16,k0+24);
    // lane>=16 holds the +8 halves (per 16-bit A 16x32 ISA layout).
    v16h afrag[2];
#pragma unroll
    for (int mi = 0; mi < 2; ++mi) {
      int rA = wr * 32 + mi * 16 + l16;
      const _Float16* pa = &ldsA[rA * LDS_STRIDE + k0 + half * 8];
      v8h lo = *(const v8h*)pa;
      v8h hi = *(const v8h*)(pa + 16);
      afrag[mi] = __builtin_shufflevector(lo, hi,
          0, 1, 2, 3, 4, 5, 6, 7, 8, 9, 10, 11, 12, 13, 14, 15);
    }

#pragma unroll
    for (int ni = 0; ni < 4; ++ni) {
      // B fragment: lane holds column N=l16 -> z row (rowJ+...) with
      // K in [k0+half*8,+8)∪[k0+16+half*8,+8) (matching A fragment order).
      int rB = wc * 64 + ni * 16 + l16;
      const _Float16* pb = &ldsB[rB * LDS_STRIDE + k0 + half * 8];
      v8h lo = *(const v8h*)pb;
      v8h hi = *(const v8h*)(pb + 16);
      v16h bfrag = __builtin_shufflevector(lo, hi,
          0, 1, 2, 3, 4, 5, 6, 7, 8, 9, 10, 11, 12, 13, 14, 15);
#pragma unroll
      for (int mi = 0; mi < 2; ++mi) {
        acc[mi][ni] = __builtin_amdgcn_wmma_f32_16x16x32_f16(
            false, afrag[mi], false, bfrag,
            (short)0, acc[mi][ni], false, false);
      }
    }
  }

  // ---- epilogue: C layout VGPR r -> (M = 8*half + r, N = l16) ----
  // Streamed (non-temporal) stores: 268 MB written once, never re-read on
  // device; keep L2 free for the 2 MB z/sq/v tables reused by all WGs.
  const size_t covOff = (size_t)b * T_ * T_;
#pragma unroll
  for (int mi = 0; mi < 2; ++mi) {
#pragma unroll
    for (int ni = 0; ni < 4; ++ni) {
      int   gj  = rowJ + wc * 64 + ni * 16 + l16;
      float sqj = sqv[bT + gj];
      float vj  = vv[bT + gj];
      int giBase = rowI + wr * 32 + mi * 16 + half * 8;
      v8f c = acc[mi][ni];
#pragma unroll
      for (int r = 0; r < 8; ++r) {
        int   gi  = giBase + r;
        float sqi = sqv[bT + gi];
        float vi  = vv[bT + gi];
        float quad = c[r] - 0.5f * (sqi + sqj);
        float f = __expf(quad) * vi * vj;
        float y;
        if (gi == gj) { f += JITTER_; y = f + nv[bT + gi]; }
        else          { y = f; }
        size_t o = covOff + (size_t)gi * T_ + gj;
        __builtin_nontemporal_store(f, &outF[o]);
        __builtin_nontemporal_store(y, &outY[o]);
      }
    }
  }
}

// ---------------------------------------------------------------------------
extern "C" void kernel_launch(void* const* d_in, const int* in_sizes, int n_in,
                              void* d_out, int out_size, void* d_ws, size_t ws_size,
                              hipStream_t stream) {
  (void)in_sizes; (void)n_in; (void)out_size; (void)ws_size;
  const float* in = (const float*)d_in[0];

  // output layout: mean | f_cov_j | y_cov_j
  float* outMean = (float*)d_out;
  float* outF    = outMean + (size_t)B_ * T_;
  float* outY    = outF + (size_t)B_ * T_ * T_;

  // workspace layout: zh (f16, 2MB) | sq | v | noise
  char*     ws  = (char*)d_ws;
  _Float16* zh  = (_Float16*)ws;
  float*    sqv = (float*)(ws + (size_t)B_ * T_ * D_ * sizeof(_Float16));
  float*    vv  = sqv + B_ * T_;
  float*    nv  = vv + B_ * T_;

  kvv_prep_kernel<<<(B_ * T_ + 255) / 256, 256, 0, stream>>>(
      in, outMean, zh, sqv, vv, nv);

  dim3 grid(T_ / TILE, T_ / TILE, B_);
  kvv_cov_kernel<<<grid, 256, 0, stream>>>(zh, sqv, vv, nv, outF, outY);
}
